// APPNP_Net_15530601743032
// MI455X (gfx1250) — matrix operand order repaired and checked by reading, not compiled, not run on hardware
//
#include <hip/hip_runtime.h>
#include <hip/hip_bf16.h>
#include <math.h>

#define NNODES 100000
#define NEDGES 3200000
#define FDIM   512
#define HDIM   64
#define CDIM   64
#define KSTEPS 10
#define ALPHA  0.1f

typedef __attribute__((ext_vector_type(16))) __bf16 v16bf;
typedef __attribute__((ext_vector_type(8)))  __bf16 v8bf;
typedef __attribute__((ext_vector_type(8)))  float  v8f;

// ---------------- prep kernels ----------------

__global__ void k_fill_zero(float* __restrict__ p, int n) {
    int i = blockIdx.x * blockDim.x + threadIdx.x;
    if (i < n) p[i] = 0.0f;
}

__global__ void k_cvt_weights(const float* __restrict__ W1, const float* __restrict__ W2,
                              __bf16* __restrict__ w1b, __bf16* __restrict__ w2b) {
    int i = blockIdx.x * blockDim.x + threadIdx.x;
    if (i < HDIM * FDIM) w1b[i] = (__bf16)W1[i];
    if (i < CDIM * HDIM) w2b[i] = (__bf16)W2[i];
}

__global__ void k_degree(const int* __restrict__ ei, float* __restrict__ deg) {
    int e = blockIdx.x * blockDim.x + threadIdx.x;
    if (e < NEDGES) atomicAdd(&deg[ei[NEDGES + e]], 1.0f);
}

__global__ void k_dinv(const float* __restrict__ deg, float* __restrict__ dinv,
                       float* __restrict__ selfw) {
    int i = blockIdx.x * blockDim.x + threadIdx.x;
    if (i < NNODES) {
        float d = deg[i] + 1.0f;
        float r = rsqrtf(d);
        dinv[i]  = r;
        selfw[i] = r * r;
    }
}

__global__ void k_norm(const int* __restrict__ ei, const float* __restrict__ dinv,
                       float* __restrict__ nrm) {
    int e = blockIdx.x * blockDim.x + threadIdx.x;
    if (e < NEDGES) nrm[e] = dinv[ei[e]] * dinv[ei[NEDGES + e]];
}

// ---------------- fused MLP: h = relu(x@W1^T+b1); z0 = h@W2^T+b2 ----------------
// One wave computes a 16-row tile. GEMM1: 4 accumulators cover H=64, K=512 in
// steps of 32 (bf16 WMMA). Hidden tile goes to LDS as bf16 in C-layout, is
// re-read in A-layout, and GEMM2 (K=64) produces z0 written to hbuf and zbuf.

__global__ __launch_bounds__(256) void k_mlp_wmma(
    const float* __restrict__ x, const __bf16* __restrict__ w1b,
    const float* __restrict__ b1, const __bf16* __restrict__ w2b,
    const float* __restrict__ b2, float* __restrict__ hbuf, float* __restrict__ zbuf)
{
    __shared__ __bf16 hs[8][16][HDIM];   // 16 KB: per-wave 16x64 bf16 hidden tile

    const int lane = threadIdx.x & 31;
    const int wid  = threadIdx.x >> 5;
    const int rt   = blockIdx.x * 8 + wid;          // 16-row tile index
    if (rt >= NNODES / 16) return;                  // N is a multiple of 16
    const int m0    = rt * 16;
    const int col0  = lane & 15;
    const int hi    = lane >> 4;                    // lane half select
    const int koff  = hi * 8;                       // A-operand K chunk offset
    const int kboff = hi * 16;                      // unused; B loads use koff

    // ---- GEMM1: [16x512] x [512x64] ----
    v8f acc[4] = {};
    const float* xr = x + (size_t)(m0 + col0) * FDIM;

    for (int kk = 0; kk < FDIM; kk += 32) {
        const float* p = xr + kk + koff;
        float4 f0 = *(const float4*)(p);
        float4 f1 = *(const float4*)(p + 4);
        float4 f2 = *(const float4*)(p + 16);
        float4 f3 = *(const float4*)(p + 20);
        v16bf a;
        a[0]=(__bf16)f0.x;  a[1]=(__bf16)f0.y;  a[2]=(__bf16)f0.z;  a[3]=(__bf16)f0.w;
        a[4]=(__bf16)f1.x;  a[5]=(__bf16)f1.y;  a[6]=(__bf16)f1.z;  a[7]=(__bf16)f1.w;
        a[8]=(__bf16)f2.x;  a[9]=(__bf16)f2.y;  a[10]=(__bf16)f2.z; a[11]=(__bf16)f2.w;
        a[12]=(__bf16)f3.x; a[13]=(__bf16)f3.y; a[14]=(__bf16)f3.z; a[15]=(__bf16)f3.w;
        #pragma unroll
        for (int j = 0; j < 4; ++j) {
            const __bf16* bp = w1b + (size_t)(16 * j + col0) * FDIM + kk + koff;
            v16bf b = __builtin_shufflevector(*(const v8bf*)(bp), *(const v8bf*)(bp + 16), 0,1,2,3,4,5,6,7,8,9,10,11,12,13,14,15);
            acc[j] = __builtin_amdgcn_wmma_f32_16x16x32_bf16(
                false, a, false, b, (short)0, acc[j], false, false);
        }
    }

    // bias + relu, stash hidden tile in LDS (bf16, row-major 16x64)
    #pragma unroll
    for (int j = 0; j < 4; ++j) {
        float bias = b1[16 * j + col0];
        #pragma unroll
        for (int r = 0; r < 8; ++r) {
            float v = fmaxf(acc[j][r] + bias, 0.0f);
            hs[wid][r + 8 * hi][16 * j + col0] = (__bf16)v;
        }
    }
    // same-wave LDS write->read: DS ops are in-order, compiler inserts s_wait_dscnt

    // ---- GEMM2: [16x64] x [64x64] ----
    v8f d[4] = {};
    const __bf16* hp = &hs[wid][col0][0];
    #pragma unroll
    for (int kq = 0; kq < HDIM; kq += 32) {
        v8bf c1 = *(const v8bf*)(hp + kq + koff);
        v8bf c2 = *(const v8bf*)(hp + kq + koff + 16);
        v16bf a2 = __builtin_shufflevector(c1, c2, 0,1,2,3,4,5,6,7,8,9,10,11,12,13,14,15);
        #pragma unroll
        for (int j = 0; j < 4; ++j) {
            const __bf16* bq = w2b + (size_t)(16 * j + col0) * HDIM + kq + koff;
            v16bf b = __builtin_shufflevector(*(const v8bf*)(bq), *(const v8bf*)(bq + 16), 0,1,2,3,4,5,6,7,8,9,10,11,12,13,14,15);
            d[j] = __builtin_amdgcn_wmma_f32_16x16x32_bf16(
                false, a2, false, b, (short)0, d[j], false, false);
        }
    }

    #pragma unroll
    for (int j = 0; j < 4; ++j) {
        float bias = b2[16 * j + col0];
        #pragma unroll
        for (int r = 0; r < 8; ++r) {
            float v = d[j][r] + bias;
            size_t idx = (size_t)(m0 + r + 8 * hi) * CDIM + 16 * j + col0;
            hbuf[idx] = v;   // teleport term
            zbuf[idx] = v;   // propagation state
        }
    }
}

// ---------------- APPNP propagation ----------------

__global__ void k_selfinit(const float* __restrict__ selfw, const float4* __restrict__ z4,
                           float4* __restrict__ agg4) {
    long long t = (long long)blockIdx.x * blockDim.x + threadIdx.x;
    if (t < (long long)NNODES * (CDIM / 4)) {
        float s = selfw[(int)(t >> 4)];
        float4 z = z4[t];
        agg4[t] = make_float4(s * z.x, s * z.y, s * z.z, s * z.w);
    }
}

__global__ void k_scatter(const int* __restrict__ ei, const float* __restrict__ nrm,
                          const float* __restrict__ z, float* __restrict__ agg) {
    long long t = (long long)blockIdx.x * blockDim.x + threadIdx.x;
    if (t < (long long)NEDGES * CDIM) {
        int e = (int)(t >> 6);
        int c = (int)(t & 63);
        int s = ei[e];
        int dv = ei[NEDGES + e];
        atomicAdd(&agg[(size_t)dv * CDIM + c], nrm[e] * z[(size_t)s * CDIM + c]);
    }
}

__global__ void k_combine(const float4* __restrict__ agg4, const float4* __restrict__ h4,
                          float4* __restrict__ z4) {
    long long t = (long long)blockIdx.x * blockDim.x + threadIdx.x;
    if (t < (long long)NNODES * (CDIM / 4)) {
        float4 a = agg4[t];
        float4 h = h4[t];
        z4[t] = make_float4(0.9f * a.x + ALPHA * h.x, 0.9f * a.y + ALPHA * h.y,
                            0.9f * a.z + ALPHA * h.z, 0.9f * a.w + ALPHA * h.w);
    }
}

// ---------------- log_softmax over C=64, one wave32 per row ----------------

__global__ __launch_bounds__(256) void k_logsoftmax(const float* __restrict__ z,
                                                    float* __restrict__ out) {
    int row = blockIdx.x * 8 + (threadIdx.x >> 5);
    if (row >= NNODES) return;
    int lane = threadIdx.x & 31;
    const float* zr = z + (size_t)row * CDIM;
    float v0 = zr[lane], v1 = zr[lane + 32];
    float m = fmaxf(v0, v1);
    #pragma unroll
    for (int off = 16; off > 0; off >>= 1) m = fmaxf(m, __shfl_xor(m, off, 32));
    float s = __expf(v0 - m) + __expf(v1 - m);
    #pragma unroll
    for (int off = 16; off > 0; off >>= 1) s += __shfl_xor(s, off, 32);
    float lse = m + __logf(s);
    out[(size_t)row * CDIM + lane]      = v0 - lse;
    out[(size_t)row * CDIM + lane + 32] = v1 - lse;
}

// ---------------- host launcher ----------------

extern "C" void kernel_launch(void* const* d_in, const int* in_sizes, int n_in,
                              void* d_out, int out_size, void* d_ws, size_t ws_size,
                              hipStream_t stream) {
    (void)in_sizes; (void)n_in; (void)out_size; (void)ws_size;

    const float* x  = (const float*)d_in[0];
    const float* W1 = (const float*)d_in[1];
    const float* b1 = (const float*)d_in[2];
    const float* W2 = (const float*)d_in[3];
    const float* b2 = (const float*)d_in[4];
    const int*   ei = (const int*)d_in[5];   // [2, E]: row 0 = src, row 1 = dst

    char* ws = (char*)d_ws;
    size_t off = 0;
    auto alloc = [&](size_t bytes) -> void* {
        void* p = ws + off;
        off = (off + bytes + 255) & ~(size_t)255;
        return p;
    };

    float*  deg    = (float*)alloc((size_t)NNODES * 4);
    float*  dinv   = (float*)alloc((size_t)NNODES * 4);
    float*  selfw  = (float*)alloc((size_t)NNODES * 4);
    float*  nrm    = (float*)alloc((size_t)NEDGES * 4);
    __bf16* w1b    = (__bf16*)alloc((size_t)HDIM * FDIM * 2);
    __bf16* w2b    = (__bf16*)alloc((size_t)CDIM * HDIM * 2);
    float*  hbuf   = (float*)alloc((size_t)NNODES * CDIM * 4);
    float*  zbuf   = (float*)alloc((size_t)NNODES * CDIM * 4);
    float*  aggbuf = (float*)alloc((size_t)NNODES * CDIM * 4);

    // normalization
    k_fill_zero<<<(NNODES + 255) / 256, 256, 0, stream>>>(deg, NNODES);
    k_cvt_weights<<<(HDIM * FDIM + 255) / 256, 256, 0, stream>>>(W1, W2, w1b, w2b);
    k_degree<<<(NEDGES + 255) / 256, 256, 0, stream>>>(ei, deg);
    k_dinv<<<(NNODES + 255) / 256, 256, 0, stream>>>(deg, dinv, selfw);
    k_norm<<<(NEDGES + 255) / 256, 256, 0, stream>>>(ei, dinv, nrm);

    // fused MLP (WMMA): h, z0
    k_mlp_wmma<<<(NNODES / 16 + 7) / 8, 256, 0, stream>>>(x, w1b, b1, w2b, b2, hbuf, zbuf);

    // APPNP propagation
    const long long nc4 = (long long)NNODES * (CDIM / 4);
    const long long ec  = (long long)NEDGES * CDIM;
    for (int it = 0; it < KSTEPS; ++it) {
        k_selfinit<<<(unsigned)((nc4 + 255) / 256), 256, 0, stream>>>(
            selfw, (const float4*)zbuf, (float4*)aggbuf);
        k_scatter<<<(unsigned)((ec + 255) / 256), 256, 0, stream>>>(ei, nrm, zbuf, aggbuf);
        k_combine<<<(unsigned)((nc4 + 255) / 256), 256, 0, stream>>>(
            (const float4*)aggbuf, (const float4*)hbuf, (float4*)zbuf);
    }

    k_logsoftmax<<<(NNODES + 7) / 8, 256, 0, stream>>>(zbuf, (float*)d_out);
}
